// ProteinAWDLSTM_56324201120106
// MI455X (gfx1250) — matrix operand order repaired and checked by reading, not compiled, not run on hardware
//
#include <hip/hip_runtime.h>
#include <stdint.h>

// ---------------------------------------------------------------------------
// Problem constants (from reference)
// ---------------------------------------------------------------------------
#define T_SEQ 512
#define BATCH 32
#define DIN   512
#define HHID  1280
#define MROWS (T_SEQ * BATCH)   // 16384

typedef __bf16 bf16_t;
typedef __attribute__((ext_vector_type(16))) __bf16 v16bf;
typedef __attribute__((ext_vector_type(8)))  float  v8f;
typedef __attribute__((ext_vector_type(4)))  unsigned int u32x4;

union V16BF { v16bf v; u32x4 u[2]; };

// ---------------------------------------------------------------------------
// helpers
// ---------------------------------------------------------------------------
static __device__ __forceinline__ unsigned short f32_to_bf16_rne(float f) {
  unsigned u = __builtin_bit_cast(unsigned, f);
  if ((u & 0x7f800000u) == 0x7f800000u) {
    return (unsigned short)((u >> 16) | (((u & 0xffffu) != 0u) ? 0x0040u : 0u));
  }
  u += 0x7fffu + ((u >> 16) & 1u);
  return (unsigned short)(u >> 16);
}

static __device__ __forceinline__ bf16_t f2bf(float f) {
  unsigned short s = f32_to_bf16_rne(f);
  return __builtin_bit_cast(bf16_t, s);
}

static __device__ __forceinline__ float sigm(float x) {
  return 1.0f / (1.0f + __expf(-x));
}

// A-matrix fragment: 16x32 bf16, row-major source, row stride K (elements).
// CDNA5 layout: lane L (L<16) holds row m=L, K = {k0..k0+7, k0+16..k0+23};
// lane L>=16 holds row m=L-16, K = {k0+8..k0+15, k0+24..k0+31}.
static __device__ __forceinline__ v16bf load_a_frag(const bf16_t* __restrict__ A,
                                                    int K, int m0, int k0, int lane) {
  int m    = m0 + (lane & 15);
  int ksel = (lane >> 4) * 8;
  const bf16_t* p = A + (size_t)m * K + k0 + ksel;
  V16BF r;
  r.u[0] = *(const u32x4*)(p);        // 8 halfs
  r.u[1] = *(const u32x4*)(p + 16);   // 8 halfs, +16 elements
  return r.v;
}

// B-matrix fragment: 32x16 bf16 where B[k][n] = W[n0+n][k], W row-major N x K.
// Lane L holds column n = L&15; lanes 0-15: K={k0..k0+7, k0+16..k0+23}, lanes 16-31: K={k0+8..k0+15, k0+24..k0+31}.
static __device__ __forceinline__ v16bf load_b_frag(const bf16_t* __restrict__ W,
                                                    int K, int n0, int k0, int lane) {
  int n  = n0 + (lane & 15);
  int kk = k0 + (lane >> 4) * 8;
  const bf16_t* p = W + (size_t)n * K + kk;
  V16BF r;
  r.u[0] = *(const u32x4*)(p);
  r.u[1] = *(const u32x4*)(p + 16);
  return r.v;
}

static __device__ __forceinline__ v8f wmma_bf16(v16bf a, v16bf b, v8f c) {
  return __builtin_amdgcn_wmma_f32_16x16x32_bf16(false, a, false, b,
                                                 (short)0, c, false, false);
}

// ---------------------------------------------------------------------------
// f32 -> bf16 conversion (grid-stride)
// ---------------------------------------------------------------------------
__global__ void cvt_f32_bf16(const float* __restrict__ src,
                             bf16_t* __restrict__ dst, long n) {
  long i      = blockIdx.x * (long)blockDim.x + threadIdx.x;
  long stride = gridDim.x * (long)blockDim.x;
  for (; i < n; i += stride) dst[i] = f2bf(src[i]);
}

// ---------------------------------------------------------------------------
// Batched input projection: out[m,n] = sum_k A[m,k]*W[n,k] + bias[n]   (f32 out)
// Block = 256 threads = 8 waves; wave tile 16(M) x 64(N); block tile 128 x 64.
// grid = (N/64, M/128).  Ping-pong double buffering (K % 64 == 0): no
// register rotates, loads of one buffer overlap WMMAs on the other.
// ---------------------------------------------------------------------------
__global__ __launch_bounds__(256) void gemm_px(const bf16_t* __restrict__ A,
                                               const bf16_t* __restrict__ W,
                                               const float* __restrict__ bias,
                                               float* __restrict__ out,
                                               int N, int K) {
  const int lane = threadIdx.x & 31;
  const int wave = threadIdx.x >> 5;
  const int m0   = blockIdx.y * 128 + wave * 16;
  const int n0   = blockIdx.x * 64;

  v8f acc[4];
#pragma unroll
  for (int j = 0; j < 4; ++j) acc[j] = (v8f){0.f,0.f,0.f,0.f,0.f,0.f,0.f,0.f};

  v16bf a0 = load_a_frag(A, K, m0, 0, lane);
  v16bf b0[4];
#pragma unroll
  for (int j = 0; j < 4; ++j) b0[j] = load_b_frag(W, K, n0 + 16 * j, 0, lane);

  for (int k0 = 0; k0 < K; k0 += 64) {
    // fill buffer 1 (k0+32 always < K since K % 64 == 0)
    v16bf a1 = load_a_frag(A, K, m0, k0 + 32, lane);
    v16bf b1[4];
#pragma unroll
    for (int j = 0; j < 4; ++j) b1[j] = load_b_frag(W, K, n0 + 16 * j, k0 + 32, lane);
    __builtin_prefetch(W + (size_t)(n0 + (lane & 15)) * K + k0 + 64, 0, 1);

#pragma unroll
    for (int j = 0; j < 4; ++j) acc[j] = wmma_bf16(a0, b0[j], acc[j]);

    // refill buffer 0 for next iteration
    if (k0 + 64 < K) {
      a0 = load_a_frag(A, K, m0, k0 + 64, lane);
#pragma unroll
      for (int j = 0; j < 4; ++j) b0[j] = load_b_frag(W, K, n0 + 16 * j, k0 + 64, lane);
    }
#pragma unroll
    for (int j = 0; j < 4; ++j) acc[j] = wmma_bf16(a1, b1[j], acc[j]);
  }

  // C layout: VGPR r -> row m0 + r + 8*(lane>>4), col n0 + (lane&15)
  const int mrow = (lane >> 4) * 8;
  const int ncol = lane & 15;
#pragma unroll
  for (int j = 0; j < 4; ++j) {
    const int n = n0 + 16 * j + ncol;
    const float bv = bias[n];
#pragma unroll
    for (int r = 0; r < 8; ++r) {
      const int m = m0 + mrow + r;
      out[(size_t)m * N + n] = acc[j][r] + bv;
    }
  }
}

// ---------------------------------------------------------------------------
// Grid-wide barrier (monotonic counter, no reset between phases).
// s_cluster_barrier is a NOP when not dispatched as a cluster (ClusterID==0);
// the atomic counter provides the actual device-wide ordering.
// ---------------------------------------------------------------------------
static __device__ __forceinline__ void grid_barrier(unsigned* bar, unsigned target) {
  __syncthreads();
  __builtin_amdgcn_s_cluster_barrier();   // CDNA5 cluster barrier (NOP if unclustered)
  if (threadIdx.x == 0) {
    __threadfence();
    atomicAdd(bar, 1u);
    while (__hip_atomic_load(bar, __ATOMIC_ACQUIRE, __HIP_MEMORY_SCOPE_AGENT) < target) {
      __builtin_amdgcn_s_sleep(1);
    }
  }
  __syncthreads();
}

__global__ void reset_bar(unsigned* bar) { *bar = 0u; }

// ---------------------------------------------------------------------------
// Persistent recurrent kernel for one layer.
//   N = 4*H gates; grid = H/64 blocks of 256 threads (8 waves).
//   Block b owns H-slice [b*64, b*64+64) ACROSS ALL FOUR GATES:
//     wave w -> gate g = w>>1, cols g*H + b*64 + (w&1)*32 ... +32.
//   Per step:
//     1. stage h (B x H bf16) into LDS via async global->LDS  (ASYNCcnt)
//     2. WMMA: acc = h @ whT for the block's 4x64 gate columns
//     3. pre tiles (+px[t]+bh) -> LDS; gates computed block-locally,
//        c lives in LDS for the whole sequence (written out once at the end)
//     4. write h (f32 ys + bf16 state), ONE grid barrier per step.
//   Reset mask rm[t+1] (tokens[t]==0, t>=1) folded into the state store.
//   Dynamic LDS: B*H*2 (h) + 4*B*64*4 (pre) + B*64*4 (c)  <= 120 KB.
// ---------------------------------------------------------------------------
__global__ __launch_bounds__(256) void lstm_recurrent(
    const float* __restrict__ px,      // [T, B, N]
    const bf16_t* __restrict__ wh,     // [N, H] row-major (bf16)
    const float* __restrict__ bh,      // [N]
    const int* __restrict__ tokens,    // [T, B]
    bf16_t* __restrict__ hstate,       // [B, H]  (global h broadcast buffer)
    float*  __restrict__ cstate,       // [B, H]  (written once at the end)
    float*  __restrict__ ys_f32,       // [T, B, H] or nullptr
    bf16_t* __restrict__ ys_bf16,      // [T, B, H] or nullptr
    unsigned* __restrict__ bar,
    int nblocks, int N, int H) {
  extern __shared__ char smem[];
  bf16_t* lds_h   = (bf16_t*)smem;                                   // [B*H]
  float*  pre_lds = (float*)(smem + (size_t)BATCH * H * sizeof(bf16_t)); // [4*B*64]
  float*  c_lds   = pre_lds + 4 * BATCH * 64;                        // [B*64]

  const int lane     = threadIdx.x & 31;
  const int wave     = threadIdx.x >> 5;
  const int tid      = blockIdx.x * 256 + threadIdx.x;
  const int nthreads = nblocks * 256;
  const int gate     = wave >> 1;               // 0..3
  const int jbase    = (wave & 1) * 32;         // col offset inside block's 64-slice
  const int wave_n0  = gate * H + blockIdx.x * 64 + jbase;  // global gate column
  unsigned phase = 0;

  // zero initial state: global h broadcast + LDS-resident c slice
  for (int i = tid; i < BATCH * H; i += nthreads) hstate[i] = f2bf(0.0f);
  for (int i = threadIdx.x; i < BATCH * 64; i += 256) c_lds[i] = 0.0f;
  phase += (unsigned)nblocks;
  grid_barrier(bar, phase);

  const int h_chunks = (BATCH * H) / 8;   // 16-byte chunks of hstate

  for (int t = 0; t < T_SEQ; ++t) {
    // ---- 1. stage hstate -> LDS via async global->LDS (ASYNCcnt) ------
    for (int cidx = threadIdx.x; cidx < h_chunks; cidx += 256) {
      unsigned lds_byte =
          (unsigned)(uintptr_t)(lds_h + (size_t)cidx * 8);
      unsigned long long gaddr =
          (unsigned long long)(uintptr_t)(hstate + (size_t)cidx * 8);
      asm volatile("global_load_async_to_lds_b128 %0, %1, off"
                   :: "v"(lds_byte), "v"(gaddr)
                   : "memory");
    }
    asm volatile("s_wait_asynccnt 0x0" ::: "memory");
    __syncthreads();

    // ---- 2. WMMA: acc = h @ whT (wave's 32 gate columns) --------------
    v8f a00 = {0.f,0.f,0.f,0.f,0.f,0.f,0.f,0.f};
    v8f a01 = a00, a10 = a00, a11 = a00;

    v16bf b0[2];
    b0[0] = load_b_frag(wh, H, wave_n0,      0, lane);
    b0[1] = load_b_frag(wh, H, wave_n0 + 16, 0, lane);
    for (int k0 = 0; k0 < H; k0 += 64) {
      v16bf b1[2];
      b1[0] = load_b_frag(wh, H, wave_n0,      k0 + 32, lane);
      b1[1] = load_b_frag(wh, H, wave_n0 + 16, k0 + 32, lane);
      {
        v16bf fa0 = load_a_frag(lds_h, H,  0, k0, lane);
        v16bf fa1 = load_a_frag(lds_h, H, 16, k0, lane);
        a00 = wmma_bf16(fa0, b0[0], a00);
        a01 = wmma_bf16(fa0, b0[1], a01);
        a10 = wmma_bf16(fa1, b0[0], a10);
        a11 = wmma_bf16(fa1, b0[1], a11);
      }
      if (k0 + 64 < H) {
        b0[0] = load_b_frag(wh, H, wave_n0,      k0 + 64, lane);
        b0[1] = load_b_frag(wh, H, wave_n0 + 16, k0 + 64, lane);
      }
      {
        v16bf fa0 = load_a_frag(lds_h, H,  0, k0 + 32, lane);
        v16bf fa1 = load_a_frag(lds_h, H, 16, k0 + 32, lane);
        a00 = wmma_bf16(fa0, b1[0], a00);
        a01 = wmma_bf16(fa0, b1[1], a01);
        a10 = wmma_bf16(fa1, b1[0], a10);
        a11 = wmma_bf16(fa1, b1[1], a11);
      }
    }

    // ---- 3. pre (+px+bh) -> LDS exchange ------------------------------
    {
      const float* pxt = px + (size_t)t * BATCH * N;
      const int mrow = (lane >> 4) * 8;
      const int ncol = lane & 15;
      v8f accs[2][2] = {{a00, a01}, {a10, a11}};
#pragma unroll
      for (int mi = 0; mi < 2; ++mi) {
#pragma unroll
        for (int nj = 0; nj < 2; ++nj) {
          const int n  = wave_n0 + 16 * nj + ncol;          // global gate col
          const int jj = jbase + 16 * nj + ncol;            // 0..63 in slice
          const float bv = bh[n];
#pragma unroll
          for (int r = 0; r < 8; ++r) {
            const int b = 16 * mi + mrow + r;               // batch index
            pre_lds[((size_t)gate * BATCH + b) * 64 + jj] =
                accs[mi][nj][r] + pxt[(size_t)b * N + n] + bv;
          }
        }
      }
    }
    __syncthreads();

    // ---- 4. block-local gates; c stays in LDS -------------------------
    for (int idx = threadIdx.x; idx < BATCH * 64; idx += 256) {
      const int b  = idx >> 6;
      const int jj = idx & 63;
      const int jg = blockIdx.x * 64 + jj;                  // global hidden idx
      const float p_i = pre_lds[((size_t)0 * BATCH + b) * 64 + jj];
      const float p_f = pre_lds[((size_t)1 * BATCH + b) * 64 + jj];
      const float p_o = pre_lds[((size_t)2 * BATCH + b) * 64 + jj];
      const float p_g = pre_lds[((size_t)3 * BATCH + b) * 64 + jj];
      const float ig = sigm(p_i);
      const float fg = sigm(p_f);
      const float og = sigm(p_o);
      const float gg = tanhf(p_g);
      const float c_new = fg * c_lds[idx] + ig * gg;
      const float h_new = og * tanhf(c_new);
      if (ys_f32)  ys_f32[(size_t)t * BATCH * H + (size_t)b * H + jg] = h_new;
      if (ys_bf16) ys_bf16[(size_t)t * BATCH * H + (size_t)b * H + jg] = f2bf(h_new);
      // fold next step's reset mask into stored state:
      // rm[t+1] = (t+1 >= 2) && tokens[(t+1)-1][b]==0  (false past the end)
      const bool rm_next = (t >= 1) && (t + 1 < T_SEQ) &&
                           (tokens[(size_t)t * BATCH + b] == 0);
      c_lds[idx] = rm_next ? 0.0f : c_new;
      hstate[(size_t)b * H + jg] = f2bf(rm_next ? 0.0f : h_new);
    }
    phase += (unsigned)nblocks;
    grid_barrier(bar, phase);
  }

  // write the LDS-resident c slice out once (c_last for this layer)
  __syncthreads();
  for (int idx = threadIdx.x; idx < BATCH * 64; idx += 256) {
    const int b  = idx >> 6;
    const int jg = blockIdx.x * 64 + (idx & 63);
    cstate[(size_t)b * H + jg] = c_lds[idx];
  }
}

// ---------------------------------------------------------------------------
// finalize: h_last = ys2[T-1], c_last = cstate
// ---------------------------------------------------------------------------
__global__ void finalize(const float* __restrict__ ys_tail,
                         const float* __restrict__ cstate,
                         float* __restrict__ out_h,
                         float* __restrict__ out_c, int n) {
  int i = blockIdx.x * blockDim.x + threadIdx.x;
  if (i < n) {
    out_h[i] = ys_tail[i];
    out_c[i] = cstate[i];
  }
}

// ---------------------------------------------------------------------------
// launch
// ---------------------------------------------------------------------------
extern "C" void kernel_launch(void* const* d_in, const int* in_sizes, int n_in,
                              void* d_out, int out_size, void* d_ws, size_t ws_size,
                              hipStream_t stream) {
  (void)in_sizes; (void)n_in; (void)out_size; (void)ws_size;

  const float* x      = (const float*)d_in[0];
  const int*   tokens = (const int*)d_in[1];
  const float* wi[3]  = {(const float*)d_in[2], (const float*)d_in[6], (const float*)d_in[10]};
  const float* bi[3]  = {(const float*)d_in[3], (const float*)d_in[7], (const float*)d_in[11]};
  const float* wh[3]  = {(const float*)d_in[4], (const float*)d_in[8], (const float*)d_in[12]};
  const float* bh[3]  = {(const float*)d_in[5], (const float*)d_in[9], (const float*)d_in[13]};

  // layer dims
  const int din_l[3]  = {DIN, HHID, HHID};
  const int dout_l[3] = {HHID, HHID, DIN};

  // workspace carve-up (all chunks multiple of 256 B)
  char* ws = (char*)d_ws;
  float*  px_buf = (float*)ws;   ws += (size_t)MROWS * 4 * HHID * sizeof(float);   // 320 MiB
  bf16_t* act0   = (bf16_t*)ws;  ws += (size_t)MROWS * HHID * sizeof(bf16_t);      //  40 MiB
  bf16_t* act1   = (bf16_t*)ws;  ws += (size_t)MROWS * HHID * sizeof(bf16_t);      //  40 MiB
  bf16_t* wib    = (bf16_t*)ws;  ws += (size_t)4 * HHID * HHID * sizeof(bf16_t);   // 12.5 MiB
  bf16_t* whb    = (bf16_t*)ws;  ws += (size_t)4 * HHID * HHID * sizeof(bf16_t);   // 12.5 MiB
  bf16_t* hstate = (bf16_t*)ws;  ws += (size_t)BATCH * HHID * sizeof(bf16_t);
  float*  cstate = (float*)ws;   ws += (size_t)BATCH * HHID * sizeof(float);
  unsigned* bar  = (unsigned*)ws;

  float* out_f = (float*)d_out;

  bf16_t* acts[2] = {act0, act1};

  // x -> bf16 (layer-0 activation buffer, row stride DIN)
  {
    long n = (long)MROWS * DIN;
    cvt_f32_bf16<<<4096, 256, 0, stream>>>(x, act0, n);
  }

  for (int l = 0; l < 3; ++l) {
    const int Kin = din_l[l];
    const int H   = dout_l[l];
    const int N   = 4 * H;

    // weights -> bf16
    cvt_f32_bf16<<<4096, 256, 0, stream>>>(wi[l], wib, (long)N * Kin);
    cvt_f32_bf16<<<4096, 256, 0, stream>>>(wh[l], whb, (long)N * H);

    // px = act @ wiT + bi   (f32)
    dim3 grid(N / 64, MROWS / 128);
    gemm_px<<<grid, 256, 0, stream>>>(acts[l & 1], wib, bi[l], px_buf, N, Kin);

    // recurrence (persistent, grid-barriered; c lives in LDS)
    reset_bar<<<1, 1, 0, stream>>>(bar);
    const int nb = H / 64;   // 20, 20, 8
    float*  ys_f = (l == 2) ? out_f : nullptr;
    bf16_t* ys_b = (l == 2) ? nullptr : acts[(l + 1) & 1];
    const size_t smem = (size_t)BATCH * H * sizeof(bf16_t)      // h stage
                      + (size_t)4 * BATCH * 64 * sizeof(float)  // pre exchange
                      + (size_t)BATCH * 64 * sizeof(float);     // persistent c
    lstm_recurrent<<<nb, 256, smem, stream>>>(px_buf, whb, bh[l], tokens,
                                              hstate, cstate,
                                              ys_f, ys_b, bar, nb, N, H);
  }

  // h_last = ys2[T-1], c_last = cstate   (both [B, DIN])
  {
    const int n = BATCH * DIN;
    const float* ys_tail = out_f + (size_t)(T_SEQ - 1) * BATCH * DIN;
    float* out_h = out_f + (size_t)T_SEQ * BATCH * DIN;
    float* out_c = out_h + n;
    finalize<<<(n + 255) / 256, 256, 0, stream>>>(ys_tail, cstate, out_h, out_c, n);
  }
}
